// DocumentGraph_75187697484282
// MI455X (gfx1250) — matrix operand order repaired and checked by reading, not compiled, not run on hardware
//
#include <hip/hip_runtime.h>
#include <math.h>

// ---------------- constants from the reference ----------------
#define B_    32
#define N_    1024
#define E_    512
#define F0_   300      // logical feature size
#define FP_   320      // padded feature size (multiple of 64)
#define CP_   640      // padded concat width
#define NEGBIG -9e15f

typedef __attribute__((ext_vector_type(16))) __bf16 v16bf;
typedef __attribute__((ext_vector_type(8)))  float  v8f;

// =====================================================================
// Batched mixed-precision GEMM: C = act(A * B + bias)   -- FULLY BRANCHLESS
// Contract (guaranteed by launcher + padding): M%64==0, Nout%64==0, K%32==0,
// lda/ldb/ldc %4==0, all base pointers 16B aligned.
//  A: fp32 row-major [M,K];  B: fp32 [K,N] (BT=0) or [N,K] (BT=1)
// bf16 WMMA 16x16x32, fp32 accumulate.
// Block = 128 threads = 4 waves.  Block tile = 64(M) x 64(N):
//   wave w owns rows m0+w*16; 4 accumulators over n-tiles sharing one A frag.
//   B panel (32K x 64N) staged in DOUBLE-BUFFERED LDS as bf16
//   (column-major, K-stride 40) -> one s_barrier per k-step.
// Fragment layouts used here (contiguous k order, same for A and B; wave32):
//  A 16x32 bf16 : lane l -> row M=l&15, kh=l>>4 ; elems i = K[kh*16 + i]
//  B 32x16 bf16 : lane l -> col N=l&15 ; elems i = K[kh*16 + i]
//  C 16x16 f32  : elem r -> M = r + 8*kh, N = l&15
// =====================================================================

#define KSTRIDE 40   // bf16 per n-row in LDS (32 data + 8 pad, 80B: 16B-aligned)

__device__ __forceinline__ v16bf loadBfrag(const __bf16* __restrict__ Bs,
                                           int nt, int l15, int kh)
{
    const __bf16* p = Bs + (nt * 16 + l15) * KSTRIDE + kh * 16;  // 16B aligned
    union { int4 q[2]; v16bf v; } u;
    u.q[0] = *(const int4*)(p);
    u.q[1] = *(const int4*)(p + 8);
    return u.v;
}

template <int BT, int ACT>
__global__ void k_wmma_gemm(const float* __restrict__ A, long sA, int lda,
                            const float* __restrict__ Bm, long sB, int ldb,
                            const float* __restrict__ bias,
                            float* __restrict__ C, long sC, int ldc,
                            int Nout, int K)
{
    __shared__ __bf16 Bs[2][64 * KSTRIDE];

    const int tid  = threadIdx.x;
    const int wave = tid >> 5;
    const int lane = tid & 31;
    const int l15  = lane & 15;
    const int kh   = lane >> 4;

    const int m0    = blockIdx.x * 64 + wave * 16;
    const int nbase = blockIdx.y * 64;

    const float* Ab = A  + (long)blockIdx.z * sA;
    const float* Bb = Bm + (long)blockIdx.z * sB;
    float*       Cb = C  + (long)blockIdx.z * sC;

    const float* arow = Ab + (long)(m0 + l15) * lda;   // A row this lane feeds

    v8f acc0 = {}, acc1 = {}, acc2 = {}, acc3 = {};

    int buf = 0;
    for (int k0 = 0; k0 < K; k0 += 32, buf ^= 1) {
        __bf16* bs = Bs[buf];
        // ---------- stage B panel (32K x 64N) into LDS, bf16 ----------
        if (BT == 0) {
            // thread t: k-row = k0+(t&31), 16 contiguous cols at nbase+(t>>5)*16
            const int kl  = tid & 31;
            const int nl0 = (tid >> 5) * 16;
            const float* r = Bb + (long)(k0 + kl) * ldb + nbase + nl0;
            const float4 f0 = *(const float4*)(r);
            const float4 f1 = *(const float4*)(r + 4);
            const float4 f2 = *(const float4*)(r + 8);
            const float4 f3 = *(const float4*)(r + 12);
            __bf16* d = bs + nl0 * KSTRIDE + kl;
            d[0 * KSTRIDE]  = (__bf16)f0.x; d[1 * KSTRIDE]  = (__bf16)f0.y;
            d[2 * KSTRIDE]  = (__bf16)f0.z; d[3 * KSTRIDE]  = (__bf16)f0.w;
            d[4 * KSTRIDE]  = (__bf16)f1.x; d[5 * KSTRIDE]  = (__bf16)f1.y;
            d[6 * KSTRIDE]  = (__bf16)f1.z; d[7 * KSTRIDE]  = (__bf16)f1.w;
            d[8 * KSTRIDE]  = (__bf16)f2.x; d[9 * KSTRIDE]  = (__bf16)f2.y;
            d[10 * KSTRIDE] = (__bf16)f2.z; d[11 * KSTRIDE] = (__bf16)f2.w;
            d[12 * KSTRIDE] = (__bf16)f3.x; d[13 * KSTRIDE] = (__bf16)f3.y;
            d[14 * KSTRIDE] = (__bf16)f3.z; d[15 * KSTRIDE] = (__bf16)f3.w;
        } else {
            // thread t: n-row = nbase+(t&63), 16 contiguous k at k0+(t>>6)*16
            const int nl = tid & 63;
            const int kq = (tid >> 6) * 16;
            const float* r = Bb + (long)(nbase + nl) * ldb + k0 + kq;
            const float4 f0 = *(const float4*)(r);
            const float4 f1 = *(const float4*)(r + 4);
            const float4 f2 = *(const float4*)(r + 8);
            const float4 f3 = *(const float4*)(r + 12);
            union { __bf16 h[16]; int4 q[2]; } pk;
            pk.h[0]  = (__bf16)f0.x; pk.h[1]  = (__bf16)f0.y;
            pk.h[2]  = (__bf16)f0.z; pk.h[3]  = (__bf16)f0.w;
            pk.h[4]  = (__bf16)f1.x; pk.h[5]  = (__bf16)f1.y;
            pk.h[6]  = (__bf16)f1.z; pk.h[7]  = (__bf16)f1.w;
            pk.h[8]  = (__bf16)f2.x; pk.h[9]  = (__bf16)f2.y;
            pk.h[10] = (__bf16)f2.z; pk.h[11] = (__bf16)f2.w;
            pk.h[12] = (__bf16)f3.x; pk.h[13] = (__bf16)f3.y;
            pk.h[14] = (__bf16)f3.z; pk.h[15] = (__bf16)f3.w;
            int4* d = (int4*)(bs + nl * KSTRIDE + kq);   // 16B aligned
            d[0] = pk.q[0];
            d[1] = pk.q[1];
        }
        __syncthreads();   // only barrier per k-step (double-buffered panel)

        // ---------- A fragment: 4x b128, unconditional ----------
        __builtin_prefetch(arow + k0 + 128, 0, 1);       // global_prefetch_b8
        const float* p = arow + k0 + kh * 16;
        const float4 c0 = *(const float4*)(p);
        const float4 c1 = *(const float4*)(p + 4);
        const float4 c2 = *(const float4*)(p + 8);
        const float4 c3 = *(const float4*)(p + 12);
        v16bf afr;
        afr[0]  = (__bf16)c0.x; afr[1]  = (__bf16)c0.y;
        afr[2]  = (__bf16)c0.z; afr[3]  = (__bf16)c0.w;
        afr[4]  = (__bf16)c1.x; afr[5]  = (__bf16)c1.y;
        afr[6]  = (__bf16)c1.z; afr[7]  = (__bf16)c1.w;
        afr[8]  = (__bf16)c2.x; afr[9]  = (__bf16)c2.y;
        afr[10] = (__bf16)c2.z; afr[11] = (__bf16)c2.w;
        afr[12] = (__bf16)c3.x; afr[13] = (__bf16)c3.y;
        afr[14] = (__bf16)c3.z; afr[15] = (__bf16)c3.w;

        const v16bf b0 = loadBfrag(bs, 0, l15, kh);
        const v16bf b1 = loadBfrag(bs, 1, l15, kh);
        const v16bf b2 = loadBfrag(bs, 2, l15, kh);
        const v16bf b3 = loadBfrag(bs, 3, l15, kh);
        acc0 = __builtin_amdgcn_wmma_f32_16x16x32_bf16(false, afr, false, b0, (short)0, acc0, false, false);
        acc1 = __builtin_amdgcn_wmma_f32_16x16x32_bf16(false, afr, false, b1, (short)0, acc1, false, false);
        acc2 = __builtin_amdgcn_wmma_f32_16x16x32_bf16(false, afr, false, b2, (short)0, acc2, false, false);
        acc3 = __builtin_amdgcn_wmma_f32_16x16x32_bf16(false, afr, false, b3, (short)0, acc3, false, false);
    }

    // ---------- epilogue: unguarded (padded C) ----------
#pragma unroll
    for (int nt = 0; nt < 4; ++nt) {
        const int cn = nbase + nt * 16 + l15;
        const float bv = bias ? bias[cn] : 0.0f;
        const v8f acc = (nt == 0) ? acc0 : (nt == 1) ? acc1 : (nt == 2) ? acc2 : acc3;
#pragma unroll
        for (int r = 0; r < 8; ++r) {
            float v = acc[r] + bv;
            if (ACT == 1) v = (v > 0.0f) ? v : (expf(v) - 1.0f);   // ELU (0 -> 0)
            Cb[(long)(m0 + r + kh * 8) * ldc + cn] = v;
        }
    }
}

// ------------- weight/bias pad-copies into zero-padded workspace -------------
// dst [320 x 320] from src [Ks x Ns] (row-major), zero pad
__global__ void k_pad_mat(const float* __restrict__ src, float* __restrict__ dst,
                          int Ks, int Ns)
{
    int i = blockIdx.x * blockDim.x + threadIdx.x;
    if (i >= FP_ * FP_) return;
    int k = i / FP_, n = i - k * FP_;
    dst[i] = (k < Ks && n < Ns) ? src[(long)k * Ns + n] : 0.0f;
}
// A1 [300 x 600] -> dst [320 x 640] with k' = k<300 ? k : k+20 (concat gap)
__global__ void k_pad_A1(const float* __restrict__ src, float* __restrict__ dst)
{
    int i = blockIdx.x * blockDim.x + threadIdx.x;
    if (i >= FP_ * CP_) return;
    int n = i / CP_, k = i - n * CP_;
    float v = 0.0f;
    if (n < F0_) {
        if (k < F0_)                      v = src[(long)n * 600 + k];
        else if (k >= FP_ && k < FP_+F0_) v = src[(long)n * 600 + (k - (FP_-F0_))];
    }
    dst[i] = v;
}
__global__ void k_pad_vec(const float* __restrict__ src, float* __restrict__ dst,
                          int n, int np)
{
    int i = blockIdx.x * blockDim.x + threadIdx.x;
    if (i < np) dst[i] = (i < n) ? src[i] : 0.0f;
}

// ---------- embedding gather into padded rows: hidden[.,0:300]=emb, pad=0 ----
__global__ void k_gather(const int* __restrict__ idx, const float* __restrict__ emb,
                         float* __restrict__ out, long total)
{
    long i = (long)blockIdx.x * blockDim.x + threadIdx.x;
    if (i >= total) return;
    long row = i / FP_; int f = (int)(i - row * FP_);
    out[i] = (f < F0_) ? emb[(long)idx[row] * F0_ + f] : 0.0f;
}

// tiny dot: out[0] = dot(a,b,K)  (single block)
__global__ void k_dot_scalar(const float* __restrict__ a, const float* __restrict__ b,
                             float* __restrict__ out, int K)
{
    __shared__ float red[256];
    float s = 0.0f;
    for (int k = threadIdx.x; k < K; k += blockDim.x) s += a[k] * b[k];
    red[threadIdx.x] = s; __syncthreads();
    for (int off = blockDim.x >> 1; off > 0; off >>= 1) {
        if ((int)threadIdx.x < off) red[threadIdx.x] += red[threadIdx.x + off];
        __syncthreads();
    }
    if (threadIdx.x == 0) out[0] = red[0];
}

// out[row] = maybe_leaky( dot(X[row,:K],v) + (*c0 if c0) ),  X leading dim ldx
__global__ void k_rowdot(const float* __restrict__ X, int ldx,
                         const float* __restrict__ v, const float* __restrict__ c0,
                         float* __restrict__ out, int K, float alpha)
{
    __shared__ float red[256];
    const float* x = X + (long)blockIdx.x * ldx;
    float s = 0.0f;
    for (int k = threadIdx.x; k < K; k += blockDim.x) s += x[k] * v[k];
    red[threadIdx.x] = s; __syncthreads();
    for (int off = blockDim.x >> 1; off > 0; off >>= 1) {
        if ((int)threadIdx.x < off) red[threadIdx.x] += red[threadIdx.x + off];
        __syncthreads();
    }
    if (threadIdx.x == 0) {
        float r = red[0] + (c0 ? c0[0] : 0.0f);
        if (alpha > 0.0f) r = (r > 0.0f) ? r : alpha * r;   // leaky relu
        out[blockIdx.x] = r;
    }
}

// att_e[b,e,:] = softmax_n( mask ? s1[b,n] : NEG )   block per (b,e)
__global__ void k_softmax_edge(const float* __restrict__ s1, const float* __restrict__ HT,
                               float* __restrict__ att, int Eo, int No)
{
    __shared__ float red[256];
    const int be = blockIdx.x;
    const int b  = be / Eo;
    const float* ht = HT + (long)be * No;
    const float* sr = s1 + (long)b * No;
    float m = NEGBIG;
    for (int n = threadIdx.x; n < No; n += blockDim.x) {
        float v = (ht[n] > 0.0f) ? sr[n] : NEGBIG;
        m = fmaxf(m, v);
    }
    red[threadIdx.x] = m; __syncthreads();
    for (int off = blockDim.x >> 1; off > 0; off >>= 1) {
        if ((int)threadIdx.x < off) red[threadIdx.x] = fmaxf(red[threadIdx.x], red[threadIdx.x + off]);
        __syncthreads();
    }
    m = red[0]; __syncthreads();
    float s = 0.0f;
    for (int n = threadIdx.x; n < No; n += blockDim.x) {
        float v = (ht[n] > 0.0f) ? sr[n] : NEGBIG;
        s += expf(v - m);
    }
    red[threadIdx.x] = s; __syncthreads();
    for (int off = blockDim.x >> 1; off > 0; off >>= 1) {
        if ((int)threadIdx.x < off) red[threadIdx.x] += red[threadIdx.x + off];
        __syncthreads();
    }
    const float inv = 1.0f / red[0];
    float* a = att + (long)be * No;
    for (int n = threadIdx.x; n < No; n += blockDim.x) {
        float v = (ht[n] > 0.0f) ? sr[n] : NEGBIG;
        a[n] = expf(v - m) * inv;
    }
}

// att_n[b,n,:] = softmax_e( mask ? leaky(sn[b,n]+se[b,e]) : NEG )  block per (b,n)
__global__ void k_softmax_node(const float* __restrict__ sn, const float* __restrict__ se,
                               const float* __restrict__ HT, float* __restrict__ att,
                               int No, int Eo, float alpha)
{
    __shared__ float red[256];
    const int bn = blockIdx.x;
    const int b  = bn / No;
    const int n  = bn - b * No;
    const float snv = sn[bn];
    const float* seb = se + (long)b * Eo;
    const float* htb = HT + ((long)b * Eo) * No + n;      // stride No over e
    float m = NEGBIG;
    for (int e = threadIdx.x; e < Eo; e += blockDim.x) {
        float r = snv + seb[e];
        r = (r > 0.0f) ? r : alpha * r;
        float v = (htb[(long)e * No] > 0.0f) ? r : NEGBIG;
        m = fmaxf(m, v);
    }
    red[threadIdx.x] = m; __syncthreads();
    for (int off = blockDim.x >> 1; off > 0; off >>= 1) {
        if ((int)threadIdx.x < off) red[threadIdx.x] = fmaxf(red[threadIdx.x], red[threadIdx.x + off]);
        __syncthreads();
    }
    m = red[0]; __syncthreads();
    float s = 0.0f;
    for (int e = threadIdx.x; e < Eo; e += blockDim.x) {
        float r = snv + seb[e];
        r = (r > 0.0f) ? r : alpha * r;
        float v = (htb[(long)e * No] > 0.0f) ? r : NEGBIG;
        s += expf(v - m);
    }
    red[threadIdx.x] = s; __syncthreads();
    for (int off = blockDim.x >> 1; off > 0; off >>= 1) {
        if ((int)threadIdx.x < off) red[threadIdx.x] += red[threadIdx.x + off];
        __syncthreads();
    }
    const float inv = 1.0f / red[0];
    float* a = att + (long)bn * Eo;
    for (int e = threadIdx.x; e < Eo; e += blockDim.x) {
        float r = snv + seb[e];
        r = (r > 0.0f) ? r : alpha * r;
        float v = (htb[(long)e * No] > 0.0f) ? r : NEGBIG;
        a[e] = expf(v - m) * inv;
    }
}

// comb row = [proj_row(320) | sem_row(320)]  (pads already zero)
__global__ void k_concat(const float* __restrict__ a, const float* __restrict__ b,
                         float* __restrict__ out, long total)
{
    long i = (long)blockIdx.x * blockDim.x + threadIdx.x;
    if (i >= total) return;
    long r = i / FP_; int f = (int)(i - r * FP_);
    out[r * CP_ + f]       = a[i];
    out[r * CP_ + FP_ + f] = b[i];
}

// in-place: h = LayerNorm(tanh(h)) * g + bb  over K=300, row stride ldh
__global__ void k_tanh_ln(float* __restrict__ h, int ldh, const float* __restrict__ g,
                          const float* __restrict__ bb, int K)
{
    __shared__ float buf[304];
    __shared__ float red[256];
    float* x = h + (long)blockIdx.x * ldh;
    float s = 0.0f;
    for (int k = threadIdx.x; k < K; k += blockDim.x) {
        float t = tanhf(x[k]); buf[k] = t; s += t;
    }
    red[threadIdx.x] = s; __syncthreads();
    for (int off = blockDim.x >> 1; off > 0; off >>= 1) {
        if ((int)threadIdx.x < off) red[threadIdx.x] += red[threadIdx.x + off];
        __syncthreads();
    }
    const float mean = red[0] / (float)K; __syncthreads();
    float s2 = 0.0f;
    for (int k = threadIdx.x; k < K; k += blockDim.x) {
        float d = buf[k] - mean; s2 += d * d;
    }
    red[threadIdx.x] = s2; __syncthreads();
    for (int off = blockDim.x >> 1; off > 0; off >>= 1) {
        if ((int)threadIdx.x < off) red[threadIdx.x] += red[threadIdx.x + off];
        __syncthreads();
    }
    const float rstd = rsqrtf(red[0] / (float)K + 1e-5f); __syncthreads();
    for (int k = threadIdx.x; k < K; k += blockDim.x)
        x[k] = (buf[k] - mean) * rstd * g[k] + bb[k];
}

// gate = sigmoid(dot(h_row,A2)+b2); out_row(300) = gate*sem + (1-gate)*proj
__global__ void k_gate(const float* __restrict__ h, int ldh,
                       const float* __restrict__ A2, const float* __restrict__ b2,
                       const float* __restrict__ sem, const float* __restrict__ proj,
                       int ldio, float* __restrict__ out, int K)
{
    __shared__ float red[256];
    __shared__ float gsh;
    const long hb = (long)blockIdx.x * ldh;
    const long ib = (long)blockIdx.x * ldio;
    const long ob = (long)blockIdx.x * K;
    float s = 0.0f;
    for (int k = threadIdx.x; k < K; k += blockDim.x) s += h[hb + k] * A2[k];
    red[threadIdx.x] = s; __syncthreads();
    for (int off = blockDim.x >> 1; off > 0; off >>= 1) {
        if ((int)threadIdx.x < off) red[threadIdx.x] += red[threadIdx.x + off];
        __syncthreads();
    }
    if (threadIdx.x == 0) gsh = 1.0f / (1.0f + expf(-(red[0] + b2[0])));
    __syncthreads();
    const float g = gsh;
    for (int k = threadIdx.x; k < K; k += blockDim.x)
        out[ob + k] = g * sem[ib + k] + (1.0f - g) * proj[ib + k];
}

// =====================================================================
static void gemm(hipStream_t s, const float* A, long sA, int lda,
                 const float* Bm, long sB, int ldb, int btrans,
                 const float* bias, float* C, long sC, int ldc,
                 int M, int N, int K, int act, int batch)
{
    dim3 g(M / 64, N / 64, batch), b(128, 1, 1);   // M%64==0, N%64==0
    if (btrans == 0 && act == 0)
        k_wmma_gemm<0,0><<<g, b, 0, s>>>(A, sA, lda, Bm, sB, ldb, bias, C, sC, ldc, N, K);
    else if (btrans == 0 && act == 1)
        k_wmma_gemm<0,1><<<g, b, 0, s>>>(A, sA, lda, Bm, sB, ldb, bias, C, sC, ldc, N, K);
    else
        k_wmma_gemm<1,0><<<g, b, 0, s>>>(A, sA, lda, Bm, sB, ldb, bias, C, sC, ldc, N, K);
}

extern "C" void kernel_launch(void* const* d_in, const int* in_sizes, int n_in,
                              void* d_out, int out_size, void* d_ws, size_t ws_size,
                              hipStream_t stream)
{
    (void)in_sizes; (void)n_in; (void)out_size; (void)ws_size;
    const int*   inputs = (const int*)  d_in[0];
    const float* HT     = (const float*)d_in[1];
    const float* emb    = (const float*)d_in[2];
    const float* Wp     = (const float*)d_in[3];
    const float* bp     = (const float*)d_in[4];
    const float* w2_1   = (const float*)d_in[5];
    const float* w3_1   = (const float*)d_in[6];
    const float* wc1    = (const float*)d_in[7];
    const float* a11    = (const float*)d_in[8];
    const float* a21    = (const float*)d_in[9];
    const float* W2t    = (const float*)d_in[10];
    const float* w2_2   = (const float*)d_in[11];
    const float* w3_2   = (const float*)d_in[12];
    const float* wc2    = (const float*)d_in[13];
    const float* a12    = (const float*)d_in[14];
    const float* a22    = (const float*)d_in[15];
    const float* A1     = (const float*)d_in[16];
    const float* b1     = (const float*)d_in[17];
    const float* ln_g   = (const float*)d_in[18];
    const float* ln_b   = (const float*)d_in[19];
    const float* A2     = (const float*)d_in[20];
    const float* b2     = (const float*)d_in[21];
    float* out = (float*)d_out;

    // ---------------- workspace layout (floats, padded) ----------------
    float* ws = (float*)d_ws;
    size_t off = 0;
    auto alloc = [&](size_t n) { float* p = ws + off; off += n; return p; };
    const long BN = (long)B_ * N_;          // 32768 rows
    const long BE = (long)B_ * E_;          // 16384 rows
    float* hidden = alloc(BN * FP_);        // later reused as sem
    float* proj   = alloc(BN * FP_);
    float* x4     = alloc(BN * FP_);        // later reused as h
    float* x1     = alloc(BN * FP_);
    float* xw     = alloc(BN * FP_);        // gat2 transferred x
    float* att    = alloc((size_t)BN * CP_);// att (B*E*N=16.8M) & comb (B*N*640)
    float* edge   = alloc(BE * FP_);
    float* e4     = alloc(BE * FP_);
    float* WpP    = alloc(FP_ * FP_);
    float* w21P   = alloc(FP_ * FP_);
    float* w31P   = alloc(FP_ * FP_);
    float* WtP    = alloc(FP_ * FP_);
    float* w22P   = alloc(FP_ * FP_);
    float* w32P   = alloc(FP_ * FP_);
    float* A1P    = alloc(FP_ * CP_);
    float* bpP    = alloc(FP_);
    float* b1P    = alloc(FP_);
    float* s1     = alloc(BN);
    float* sn     = alloc(BN);
    float* se     = alloc(BE);
    float* c0     = alloc(4);
    float* sem    = hidden;                 // reuse
    float* hbuf   = x4;                     // reuse

    const dim3 t256(256, 1, 1);
    const unsigned gm = (FP_ * FP_ + 255) / 256;

    // padded weight/bias copies (zero-fill pads; zeros then flow through GEMMs)
    k_pad_mat<<<gm, t256, 0, stream>>>(Wp,   WpP,  F0_, F0_);
    k_pad_mat<<<gm, t256, 0, stream>>>(w2_1, w21P, F0_, F0_);
    k_pad_mat<<<gm, t256, 0, stream>>>(w3_1, w31P, F0_, F0_);
    k_pad_mat<<<gm, t256, 0, stream>>>(W2t,  WtP,  F0_, F0_);
    k_pad_mat<<<gm, t256, 0, stream>>>(w2_2, w22P, F0_, F0_);
    k_pad_mat<<<gm, t256, 0, stream>>>(w3_2, w32P, F0_, F0_);
    k_pad_A1 <<<(FP_ * CP_ + 255) / 256, t256, 0, stream>>>(A1, A1P);
    k_pad_vec<<<2, t256, 0, stream>>>(bp, bpP, F0_, FP_);
    k_pad_vec<<<2, t256, 0, stream>>>(b1, b1P, F0_, FP_);

    // hidden = emb[inputs]  (padded rows)                 [B,N,320]
    {
        long tot = BN * FP_;
        k_gather<<<dim3((unsigned)((tot + 255) / 256)), t256, 0, stream>>>(
            inputs, emb, hidden, tot);
    }
    // proj = hidden @ Wp.T + bp
    gemm(stream, hidden, 0, FP_, WpP, 0, FP_, 1, bpP, proj, 0, FP_,
         (int)BN, FP_, FP_, 0, 1);

    // ================= gat1 (transfer=False, concat=True, alpha=.1) =========
    gemm(stream, hidden, 0, FP_, w21P, 0, FP_, 0, nullptr, x4, 0, FP_,
         (int)BN, FP_, FP_, 0, 1);                            // x4 = hidden@w2_1
    k_dot_scalar<<<1, t256, 0, stream>>>(wc1, a11, c0, F0_);
    k_rowdot<<<dim3((unsigned)BN), t256, 0, stream>>>(x4, FP_, a11 + F0_, c0, s1, F0_, 0.1f);
    k_softmax_edge<<<dim3((unsigned)BE), t256, 0, stream>>>(s1, HT, att, E_, N_);
    gemm(stream, att, (long)E_ * N_, N_, hidden, (long)N_ * FP_, FP_, 0, nullptr,
         edge, (long)E_ * FP_, FP_, E_, FP_, N_, 0, B_);       // edge = att_e @ hidden
    gemm(stream, edge, 0, FP_, w31P, 0, FP_, 0, nullptr, e4, 0, FP_,
         (int)BE, FP_, FP_, 0, 1);                            // e4 = edge@w3_1
    k_rowdot<<<dim3((unsigned)BN), t256, 0, stream>>>(x4, FP_, a21, nullptr, sn, F0_, 0.0f);
    k_rowdot<<<dim3((unsigned)BE), t256, 0, stream>>>(e4, FP_, a21 + F0_, nullptr, se, F0_, 0.0f);
    k_softmax_node<<<dim3((unsigned)BN), t256, 0, stream>>>(sn, se, HT, att, N_, E_, 0.1f);
    gemm(stream, att, (long)N_ * E_, E_, edge, (long)E_ * FP_, FP_, 0, nullptr,
         x1, (long)N_ * FP_, FP_, N_, FP_, E_, 1, B_);         // x1 = elu(att_n@edge)

    // ================= gat2 (transfer=True, concat=False, alpha=.2) =========
    gemm(stream, x1, 0, FP_, w22P, 0, FP_, 0, nullptr, x4, 0, FP_,
         (int)BN, FP_, FP_, 0, 1);                            // x4 = x1@w2_2
    gemm(stream, x1, 0, FP_, WtP, 0, FP_, 0, nullptr, xw, 0, FP_,
         (int)BN, FP_, FP_, 0, 1);                            // xw = x1@W2t
    k_dot_scalar<<<1, t256, 0, stream>>>(wc2, a12, c0, F0_);
    k_rowdot<<<dim3((unsigned)BN), t256, 0, stream>>>(x4, FP_, a12 + F0_, c0, s1, F0_, 0.2f);
    k_softmax_edge<<<dim3((unsigned)BE), t256, 0, stream>>>(s1, HT, att, E_, N_);
    gemm(stream, att, (long)E_ * N_, N_, xw, (long)N_ * FP_, FP_, 0, nullptr,
         edge, (long)E_ * FP_, FP_, E_, FP_, N_, 0, B_);       // edge = att_e @ xw
    gemm(stream, edge, 0, FP_, w32P, 0, FP_, 0, nullptr, e4, 0, FP_,
         (int)BE, FP_, FP_, 0, 1);                            // e4 = edge@w3_2
    k_rowdot<<<dim3((unsigned)BN), t256, 0, stream>>>(x4, FP_, a22, nullptr, sn, F0_, 0.0f);
    k_rowdot<<<dim3((unsigned)BE), t256, 0, stream>>>(e4, FP_, a22 + F0_, nullptr, se, F0_, 0.0f);
    k_softmax_node<<<dim3((unsigned)BN), t256, 0, stream>>>(sn, se, HT, att, N_, E_, 0.2f);
    gemm(stream, att, (long)N_ * E_, E_, edge, (long)E_ * FP_, FP_, 0, nullptr,
         sem, (long)N_ * FP_, FP_, N_, FP_, E_, 0, B_);        // sem = att_n @ edge

    // ================= output head =================
    {
        long tot = BN * FP_;
        k_concat<<<dim3((unsigned)((tot + 255) / 256)), t256, 0, stream>>>(
            proj, sem, att, tot);                              // comb in att buffer
    }
    gemm(stream, att, 0, CP_, A1P, 0, CP_, 1, b1P, hbuf, 0, FP_,
         (int)BN, FP_, CP_, 0, 1);                             // h_pre = comb@A1.T+b1
    k_tanh_ln<<<dim3((unsigned)BN), t256, 0, stream>>>(hbuf, FP_, ln_g, ln_b, F0_);
    k_gate<<<dim3((unsigned)BN), t256, 0, stream>>>(hbuf, FP_, A2, b2, sem, proj, FP_, out, F0_);
}
